// DampedMultidimEMAConv_71116068487382
// MI455X (gfx1250) — matrix operand order repaired and checked by reading, not compiled, not run on hardware
//
#include <hip/hip_runtime.h>

#define B_   8
#define N_   4096
#define C_   512
#define H_   4
#define SEG  512      // N-segment per scan block
#define WARM 512      // warm-up steps (r^512 < 1e-33 -> exact in f32)

#define TM   64       // GEMM rows per block
#define TN   128      // GEMM cols per block
#define TK   32       // k-step (one f16 WMMA depth)
#define LDF  72       // f16 per LDS row: 32 hi + 32 lo + 8 pad (144B, 16B-aligned)
#define KDIM (2 * C_) // GEMM reduction depth

typedef __attribute__((ext_vector_type(16))) _Float16 v16h;
typedef __attribute__((ext_vector_type(8)))  _Float16 v8h;
typedef __attribute__((ext_vector_type(8)))  float    v8f;

// ---------------------------------------------------------------------------
// gfx1250 async global->LDS copy helpers.
// Generic pointers into the LDS aperture carry the LDS byte offset in their
// low 32 bits (ISA 10.2); the async op computes
// its LDS address as the LDS base plus the VGPR, i.e. exactly that offset.
// ---------------------------------------------------------------------------
__device__ __forceinline__ unsigned lds_off(const void* p) {
  return (unsigned)(unsigned long long)(uintptr_t)p;
}
__device__ __forceinline__ void async_cp_b128(unsigned ldsdst, const void* gsrc) {
  asm volatile("global_load_async_to_lds_b128 %0, %1, off"
               :: "v"(ldsdst), "v"(gsrc) : "memory");
}
__device__ __forceinline__ void wait_async0() {
  asm volatile("s_wait_asynccnt 0x0" ::: "memory");
}

// ---------------------------------------------------------------------------
// Kernel 1: per-channel EMA parameters.  A = eta*sig(a)*sig(b), r = 1 - sig(a)*sig(d)
// Stored transposed [H][C] so the scan kernel reads them coalesced.
// ---------------------------------------------------------------------------
__global__ void ema_params_k(const float* __restrict__ al, const float* __restrict__ dl,
                             const float* __restrict__ bl, const float* __restrict__ eta,
                             float* __restrict__ Ap, float* __restrict__ rp) {
  int i = blockIdx.x * blockDim.x + threadIdx.x;
  if (i >= C_ * H_) return;
  int c = i / H_, h = i % H_;
  float a = 1.f / (1.f + __expf(-al[i]));
  float d = 1.f / (1.f + __expf(-dl[i]));
  float b = 1.f / (1.f + __expf(-bl[i]));
  Ap[h * C_ + c] = eta[i] * a * b;
  rp[h * C_ + c] = 1.f - a * d;
}

// ---------------------------------------------------------------------------
// Kernel 1b: one-shot hi/lo f16 split of proj_w (2 MB; amortized over all
// 512 row-blocks of the GEMM).  Layout: physical row 2c = hi, 2c+1 = lo.
// ---------------------------------------------------------------------------
__global__ void wsplit_k(const float* __restrict__ pw, _Float16* __restrict__ pwhl) {
  int i = blockIdx.x * blockDim.x + threadIdx.x;
  if (i >= C_ * KDIM) return;
  int c = i / KDIM, d = i % KDIM;
  float v = pw[i];
  _Float16 h = (_Float16)v;
  pwhl[(size_t)(2 * c) * KDIM + d]     = h;
  pwhl[(size_t)(2 * c + 1) * KDIM + d] = (_Float16)(v - (float)h);
}

// ---------------------------------------------------------------------------
// Kernel 2: bidirectional EMA scan.  Replaces the K=2048 depthwise conv with
// H=4 first-order recurrences per direction (275 GFLOP -> ~1 GFLOP), with a
// WARM-step zero-state warm-up per segment (numerically exact, see analysis).
// z is emitted pre-split as f16 (hi, lo) planes -> zero conversion work in the
// GEMM hot loop, same 4 B/element as f32 (~21-bit effective mantissa).
// Layout: physical row 2*(b*N+n) = hi[2C], row 2*(b*N+n)+1 = lo[2C];
//         fwd in cols [0,C), bwd in [C,2C).
// ---------------------------------------------------------------------------
__global__ void __launch_bounds__(256)
ema_scan_k(const float* __restrict__ x, const float* __restrict__ Ap,
           const float* __restrict__ rp, _Float16* __restrict__ zhl) {
  const int c  = blockIdx.z * 256 + threadIdx.x;
  const int b  = blockIdx.x;
  const int s0 = blockIdx.y * SEG;

  float Ah[H_], rh[H_], y[H_];
#pragma unroll
  for (int h = 0; h < H_; ++h) { Ah[h] = Ap[h * C_ + c]; rh[h] = rp[h * C_ + c]; }

  const float* xb = x + (size_t)b * N_ * C_ + c;
  _Float16*    zb = zhl + (size_t)b * N_ * 2 * KDIM;

  // ---- forward (causal) ----
#pragma unroll
  for (int h = 0; h < H_; ++h) y[h] = 0.f;
  int n0 = s0 - WARM; if (n0 < 0) n0 = 0;
  for (int n = n0; n < s0 + SEG; ++n) {
    if (n + 96 < N_) __builtin_prefetch(xb + (size_t)(n + 96) * C_, 0, 1);
    float xv = xb[(size_t)n * C_];
    float acc = 0.f;
#pragma unroll
    for (int h = 0; h < H_; ++h) { y[h] = fmaf(rh[h], y[h], xv); acc = fmaf(Ah[h], y[h], acc); }
    if (n >= s0) {
      _Float16 hh = (_Float16)acc;
      size_t off = (size_t)n * 2 * KDIM;
      zb[off + c]        = hh;
      zb[off + KDIM + c] = (_Float16)(acc - (float)hh);
    }
  }

  // ---- backward (anti-causal) ----
#pragma unroll
  for (int h = 0; h < H_; ++h) y[h] = 0.f;
  int n1 = s0 + SEG + WARM; if (n1 > N_) n1 = N_;
  const int sEnd = s0 + SEG;
  for (int n = n1 - 1; n >= s0; --n) {
    if (n >= 96) __builtin_prefetch(xb + (size_t)(n - 96) * C_, 0, 1);
    float xv = xb[(size_t)n * C_];
    float acc = 0.f;
#pragma unroll
    for (int h = 0; h < H_; ++h) { y[h] = fmaf(rh[h], y[h], xv); acc = fmaf(Ah[h], y[h], acc); }
    if (n < sEnd) {
      _Float16 hh = (_Float16)acc;
      size_t off = (size_t)n * 2 * KDIM;
      zb[off + C_ + c]        = hh;
      zb[off + KDIM + C_ + c] = (_Float16)(acc - (float)hh);
    }
  }
}

// ---------------------------------------------------------------------------
// Kernel 3: projection GEMM  out[m, c] = sum_d z[m, d] * pw[c, d] + pb[c]
// M = B*N = 32768, Kdim = 2C = 1024, Ncols = C = 512.
// 8 waves -> 64x128 tile (4 row-groups x 2 col-groups, 4 16x16 subtiles/wave).
// Pre-split f16 (hi,lo) tiles staged via double-buffered async global->LDS
// b128 copies (ASYNCcnt); hot loop is pure ds_load_b128 + 12 WMMAs per k-step.
// (hi+lo)*(hi+lo) ~= hi*hi + hi*lo + lo*hi gives ~f32 accuracy.
// ---------------------------------------------------------------------------
__global__ void __launch_bounds__(256)
proj_wmma_k(const _Float16* __restrict__ zhl, const _Float16* __restrict__ pwhl,
            const float* __restrict__ pb, float* __restrict__ out) {
  __shared__ _Float16 zt[2][TM * LDF];   // 2 x 64 x 72 f16 = 18.4 KB
  __shared__ _Float16 wt[2][TN * LDF];   // 2 x 128 x 72 f16 = 36.9 KB

  const int tid  = threadIdx.x;
  const int lane = tid & 31;
  const int wv   = tid >> 5;          // wave 0..7
  const int wr   = wv >> 1;           // row group (0..3) -> 16 rows each
  const int wc   = wv & 1;            // col group (0..1) -> 64 cols each
  const int ln16 = lane & 15;
  const int half = lane >> 4;         // 0/1
  const int row0 = blockIdx.x * TM;
  const int col0 = blockIdx.y * TN;

  // stage one (z,w) f16 hi/lo tile pair for depth [kk, kk+TK) into buffer buf
  auto stage = [&](int buf, int kk) {
#pragma unroll
    for (int i = 0; i < 2; ++i) {              // z: 64 rows x 2 planes x 4 b128
      const int chunk = tid + 256 * i;         // 512 chunks
      const int pr = chunk >> 2;               // physical row 0..127
      const int m  = pr >> 1, pl = pr & 1;     // M row, hi/lo plane
      const int c8 = (chunk & 3) * 8;          // f16 offset in row
      async_cp_b128(lds_off(&zt[buf][m * LDF + pl * 32 + c8]),
                    zhl + ((size_t)2 * (row0 + m) + pl) * KDIM + kk + c8);
    }
#pragma unroll
    for (int i = 0; i < 4; ++i) {              // w: 128 rows x 2 planes x 4 b128
      const int chunk = tid + 256 * i;         // 1024 chunks
      const int pr = chunk >> 2;               // physical row 0..255
      const int n  = pr >> 1, pl = pr & 1;
      const int c8 = (chunk & 3) * 8;
      async_cp_b128(lds_off(&wt[buf][n * LDF + pl * 32 + c8]),
                    pwhl + ((size_t)2 * (col0 + n) + pl) * KDIM + kk + c8);
    }
  };

  v8f acc[4] = {};

  stage(0, 0);
  wait_async0();
  __syncthreads();

  for (int kk = 0; kk < KDIM; kk += TK) {
    const int buf = (kk >> 5) & 1;
    if (kk + TK < KDIM) stage(buf ^ 1, kk + TK);   // prefetch next tile (async)

    // A fragment: 16-bit A 16x32 layout; lane -> row, halves hold
    // K in {0..7,16..23} / {8..15,24..31} -> two aligned b128 per plane.
    v16h a_hi, a_lo;
    {
      const _Float16* zr = &zt[buf][(wr * 16 + ln16) * LDF];
      const int ka = half * 8;
      v8h h0 = *(const v8h*)(zr + ka);
      v8h h1 = *(const v8h*)(zr + ka + 16);
      v8h l0 = *(const v8h*)(zr + 32 + ka);
      v8h l1 = *(const v8h*)(zr + 32 + ka + 16);
#pragma unroll
      for (int j = 0; j < 8; ++j) {
        a_hi[j] = h0[j]; a_hi[j + 8] = h1[j];
        a_lo[j] = l0[j]; a_lo[j + 8] = l1[j];
      }
    }

    // B fragments: lane -> col, halves hold K {0..7,16..23}/{8..15,24..31}
    const int kb = half * 8;
#pragma unroll
    for (int s = 0; s < 4; ++s) {
      const _Float16* wp = &wt[buf][(wc * 64 + s * 16 + ln16) * LDF];
      v8h h0 = *(const v8h*)(wp + kb);
      v8h h1 = *(const v8h*)(wp + kb + 16);
      v8h l0 = *(const v8h*)(wp + 32 + kb);
      v8h l1 = *(const v8h*)(wp + 32 + kb + 16);
      v16h b_hi, b_lo;
#pragma unroll
      for (int j = 0; j < 8; ++j) {
        b_hi[j] = h0[j]; b_hi[j + 8] = h1[j];
        b_lo[j] = l0[j]; b_lo[j + 8] = l1[j];
      }
      acc[s] = __builtin_amdgcn_wmma_f32_16x16x32_f16(false, a_hi, false, b_hi,
                                                      (short)0, acc[s], false, false);
      acc[s] = __builtin_amdgcn_wmma_f32_16x16x32_f16(false, a_hi, false, b_lo,
                                                      (short)0, acc[s], false, false);
      acc[s] = __builtin_amdgcn_wmma_f32_16x16x32_f16(false, a_lo, false, b_hi,
                                                      (short)0, acc[s], false, false);
    }

    wait_async0();     // next tile fully landed in LDS
    __syncthreads();   // visible across all waves / reuse-safe
  }

  // epilogue: f32 C/D layout (VGPR v -> row v + 8*half, lane%16 -> col) + bias
  const int mbase = row0 + wr * 16 + half * 8;
#pragma unroll
  for (int s = 0; s < 4; ++s) {
    const int cg = col0 + wc * 64 + s * 16 + ln16;
    const float bb = pb[cg];
#pragma unroll
    for (int v = 0; v < 8; ++v) {
      out[(size_t)(mbase + v) * C_ + cg] = acc[s][v] + bb;
    }
  }
}

// ---------------------------------------------------------------------------
extern "C" void kernel_launch(void* const* d_in, const int* in_sizes, int n_in,
                              void* d_out, int out_size, void* d_ws, size_t ws_size,
                              hipStream_t stream) {
  (void)in_sizes; (void)n_in; (void)out_size; (void)ws_size;
  const float* x   = (const float*)d_in[0];
  const float* al  = (const float*)d_in[1];
  const float* dl  = (const float*)d_in[2];
  const float* bl  = (const float*)d_in[3];
  const float* eta = (const float*)d_in[4];
  const float* pw  = (const float*)d_in[5];
  const float* pb  = (const float*)d_in[6];
  float* out = (float*)d_out;

  float* ws = (float*)d_ws;
  float* Ap = ws;                    // [H][C] f32
  float* rp = ws + C_ * H_;          // [H][C] f32
  _Float16* pwhl = (_Float16*)(ws + 2 * C_ * H_);    // [2C][KDIM] f16 (hi/lo rows)
  _Float16* zhl  = pwhl + (size_t)2 * C_ * KDIM;     // [B*N*2][KDIM] f16 (128 MB)

  ema_params_k<<<(C_ * H_ + 255) / 256, 256, 0, stream>>>(al, dl, bl, eta, Ap, rp);
  wsplit_k<<<(C_ * KDIM + 255) / 256, 256, 0, stream>>>(pw, pwhl);
  ema_scan_k<<<dim3(B_, N_ / SEG, C_ / 256), 256, 0, stream>>>(x, Ap, rp, zhl);
  proj_wmma_k<<<dim3((B_ * N_) / TM, C_ / TN), 256, 0, stream>>>(zhl, pwhl, pb, out);
}
